// AugmentationLayer_12670153523277
// MI455X (gfx1250) — matrix operand order repaired and checked by reading, not compiled, not run on hardware
//
#include <hip/hip_runtime.h>
#include <math.h>

typedef __attribute__((ext_vector_type(16))) _Float16 v16h;
typedef __attribute__((ext_vector_type(8)))  float    v8f;

#define HWSZ (224 * 224)
#define NIMG 1024
#define CCH  256

// torchvision 'reflect' padding index (pad<=2, single reflection suffices here)
__device__ __forceinline__ int refl224(int i) {
    if (i < 0)    i = -i;
    if (i >= 224) i = 446 - i;   // 2*224 - 2 - i
    return i;
}

__device__ __forceinline__ float band5(int d, float w0, float w1, float w2) {
    // symmetric 5-tap band: d in [0,4] -> {w0,w1,w2,w1,w0}, else 0
    return (d == 0 || d == 4) ? w0 : (d == 1 || d == 3) ? w1 : (d == 2) ? w2 : 0.0f;
}

// ---------------------------------------------------------------------------
// 1) Per-image params + prebaked WMMA weight operands.
//    Grid: <<<1024, 32>>>, one block per image, one lane per WMMA lane.
//    prm[n*16 + {0..6}] = cos a, sin a, hc/H, wc/W, i, j, flip
//    wbh[n][lane][e]    = horizontal-pass B operand (32x16 band, f16 layout)
//    wav[n][lane][e]    = vertical-pass   A operand (16x32 band, f16 layout)
// ---------------------------------------------------------------------------
__global__ void params_kernel(const float* __restrict__ aug_u,
                              float* __restrict__ prm,
                              _Float16* __restrict__ wbh,
                              _Float16* __restrict__ wav) {
    int n    = blockIdx.x;
    int lane = threadIdx.x;            // 0..31
    const float* u = aug_u + n * 7;

    // normalized gaussian weights from sigma (recomputed per lane; cheap)
    float sigma  = 0.1f + 1.9f * u[6];
    float inv2s2 = 1.0f / (2.0f * sigma * sigma);
    float e4 = expf(-4.0f * inv2s2);
    float e1 = expf(-1.0f * inv2s2);
    float is = 1.0f / (2.0f * e4 + 2.0f * e1 + 1.0f);
    float w0 = e4 * is, w1 = e1 * is, w2 = is;

    int Nc   = lane & 15;
    int half = lane >> 4;
    _Float16* ph = wbh + (size_t)n * 512 + lane * 16;
    _Float16* pv = wav + (size_t)n * 512 + lane * 16;
#pragma unroll
    for (int e = 0; e < 16; ++e) {
        int v  = e >> 1;
        // B layout (16-bit 32x16): K = 2v (+8 for v>=4) + 8*half + (e&1); band w[K-N]
        int Kb = ((v < 4) ? (v << 1) : (16 + ((v - 4) << 1))) + (half << 3) + (e & 1);
        ph[e] = (_Float16)band5(Kb - Nc, w0, w1, w2);
        // A operand (16x32), contiguous k order matched by vblur B: K = 16*half + 2v + (e&1); w[K-M]
        int Ka = (half << 4) + (v << 1) + (e & 1);
        pv[e] = (_Float16)band5(Ka - Nc, w0, w1, w2);
    }

    if (lane == 0) {
        float area  = 224.0f * 224.0f * (0.8f + 0.2f * u[0]);
        const float lo = -0.2876820724517809f;   // log(3/4)
        const float hi =  0.2876820724517809f;   // log(4/3)
        float ratio = expf(lo + (hi - lo) * u[1]);
        float wc = fminf(fmaxf(sqrtf(area * ratio), 1.0f), 224.0f);
        float hc = fminf(fmaxf(sqrtf(area / ratio), 1.0f), 224.0f);
        float angle = u[5] * 3.14159265358979323846f;
        float* q = prm + n * 16;
        q[0] = cosf(angle);
        q[1] = sinf(angle);
        q[2] = hc * (1.0f / 224.0f);
        q[3] = wc * (1.0f / 224.0f);
        q[4] = u[2] * (224.0f - hc);
        q[5] = u[3] * (224.0f - wc);
        q[6] = (u[4] < 0.5f) ? 1.0f : 0.0f;
    }
}

// ---------------------------------------------------------------------------
// 2) Bulk copy M -> out (float4, grid-stride) with global_prefetch
// ---------------------------------------------------------------------------
__global__ void copy_kernel(const float4* __restrict__ src,
                            float4* __restrict__ dst, int total4) {
    int idx    = blockIdx.x * blockDim.x + threadIdx.x;
    int stride = gridDim.x * blockDim.x;
    for (int i = idx; i < total4; i += stride) {
        int pf = i + stride;
        if (pf < total4) __builtin_prefetch(&src[pf], 0, 0);
        dst[i] = src[i];
    }
}

// ---------------------------------------------------------------------------
// 3) Warp: inverse rotate o flip o resized-crop, bilinear sample -> f16 ws
//    196 blocks of 256 threads per image (n is block-uniform).
// ---------------------------------------------------------------------------
__global__ void warp_kernel(const float* __restrict__ Mv,
                            const int* __restrict__ cidx,
                            const float* __restrict__ prm,
                            _Float16* __restrict__ warped) {
    int blk = blockIdx.x;
    int n   = blk / 196;
    int p   = (blk % 196) * 256 + threadIdx.x;
    int y   = p / 224;
    int x   = p % 224;

    const float* q = prm + n * 16;
    float ca = q[0], sa = q[1], hcH = q[2], wcW = q[3];
    float ii = q[4], jj = q[5], flip = q[6];

    int b = n >> 7, a = n & 127;
    const float* src = Mv + (size_t)(b * CCH + cidx[a]) * HWSZ;

    const float c = 111.5f;                 // (H-1)/2
    float dy = (float)y - c, dx = (float)x - c;
    float yr =  ca * dy + sa * dx + c;
    float xr = -sa * dy + ca * dx + c;
    bool inb = (yr >= -0.5f) && (yr <= 223.5f) && (xr >= -0.5f) && (xr <= 223.5f);
    float xf = (flip > 0.5f) ? (223.0f - xr) : xr;
    float sy = (yr + 0.5f) * hcH - 0.5f + ii;
    float sx = (xf + 0.5f) * wcW - 0.5f + jj;

    float y0f = floorf(sy), x0f = floorf(sx);
    float wy = sy - y0f, wx = sx - x0f;
    int y0i = min(max((int)y0f, 0), 223);
    int x0i = min(max((int)x0f, 0), 223);
    int y1i = min(y0i + 1, 223);
    int x1i = min(x0i + 1, 223);

    float v00 = src[y0i * 224 + x0i];
    float v01 = src[y0i * 224 + x1i];
    float v10 = src[y1i * 224 + x0i];
    float v11 = src[y1i * 224 + x1i];
    float top = v00 * (1.0f - wx) + v01 * wx;
    float bot = v10 * (1.0f - wx) + v11 * wx;
    float out = top * (1.0f - wy) + bot * wy;
    out = inb ? out : 0.0f;

    warped[(size_t)n * HWSZ + p] = (_Float16)out;
}

// ---------------------------------------------------------------------------
// 4) Horizontal blur via WMMA: D[16x16] = A(data 16x32) x B(band 32x16).
//    7 waves/block (224 thr), 28 blocks/image -> n block-uniform (scalar).
// ---------------------------------------------------------------------------
__global__ void __launch_bounds__(224)
hblur_kernel(const _Float16* __restrict__ warped,
             const _Float16* __restrict__ wbh,
             _Float16* __restrict__ hblur) {
    int n    = blockIdx.x / 28;                 // image (scalar)
    int wv   = threadIdx.x >> 5;                // wave in block: 0..6
    int lane = threadIdx.x & 31;
    int tile = (blockIdx.x % 28) * 7 + wv;      // 0..195
    int y0   = (tile / 14) << 4;
    int x0   = (tile % 14) << 4;

    int Mrow = lane & 15;
    int half = lane >> 4;
    const _Float16* img = warped + (size_t)n * HWSZ;

    // prebaked band matrix: two global_load_b128 per lane
    v16h bmat = *(const v16h*)(wbh + (size_t)n * 512 + lane * 16);

    v16h amat;
#pragma unroll
    for (int e = 0; e < 16; ++e) {
        int v  = e >> 1;
        int Ka = ((v < 4) ? (v << 1) : (16 + ((v - 4) << 1))) + (half << 3) + (e & 1);
        int xx = refl224(x0 - 2 + Ka);
        amat[e] = img[(y0 + Mrow) * 224 + xx];
    }

    v8f cacc = {};
    v8f d = __builtin_amdgcn_wmma_f32_16x16x32_f16(
        false, amat, false, bmat, (short)0, cacc, false, false);

#pragma unroll
    for (int r = 0; r < 8; ++r) {
        int yy = y0 + r + (half << 3);          // D: M = r + 8*half
        hblur[(size_t)n * HWSZ + yy * 224 + x0 + Mrow] = (_Float16)d[r];
    }
}

// ---------------------------------------------------------------------------
// 5) Vertical blur via WMMA: D = A(band 16x32) x B(data 32x16),
//    fuse + 0.05*noise and scatter to out[b, channel_idx[a]].
// ---------------------------------------------------------------------------
__global__ void __launch_bounds__(224)
vblur_kernel(const _Float16* __restrict__ hblur,
             const _Float16* __restrict__ wav,
             const int* __restrict__ cidx,
             const float* __restrict__ noise,
             float* __restrict__ out) {
    int n    = blockIdx.x / 28;                 // image (scalar)
    int wv   = threadIdx.x >> 5;
    int lane = threadIdx.x & 31;
    int tile = (blockIdx.x % 28) * 7 + wv;
    int y0   = (tile / 14) << 4;
    int x0   = (tile % 14) << 4;

    int Nc   = lane & 15;
    int half = lane >> 4;
    const _Float16* img = hblur + (size_t)n * HWSZ;

    // prebaked band matrix (A operand)
    v16h amat = *(const v16h*)(wav + (size_t)n * 512 + lane * 16);

    v16h bmat;
#pragma unroll
    for (int e = 0; e < 16; ++e) {
        int v  = e >> 1;
        int Kb = (half << 4) + (v << 1) + (e & 1);
        int yy = refl224(y0 - 2 + Kb);
        bmat[e] = img[yy * 224 + x0 + Nc];
    }

    v8f cacc = {};
    v8f d = __builtin_amdgcn_wmma_f32_16x16x32_f16(
        false, amat, false, bmat, (short)0, cacc, false, false);

    int b = n >> 7, a = n & 127;
    size_t obase = (size_t)(b * CCH + cidx[a]) * HWSZ;
    size_t nbase = (size_t)n * HWSZ;
#pragma unroll
    for (int r = 0; r < 8; ++r) {
        int yy  = y0 + r + (half << 3);
        int off = yy * 224 + x0 + Nc;
        out[obase + off] = d[r] + 0.05f * noise[nbase + off];
    }
}

// ---------------------------------------------------------------------------
extern "C" void kernel_launch(void* const* d_in, const int* in_sizes, int n_in,
                              void* d_out, int out_size, void* d_ws, size_t ws_size,
                              hipStream_t stream) {
    const float* M     = (const float*)d_in[0];   // [8,256,224,224]
    const int*   cidx  = (const int*)d_in[1];     // [128]
    const float* aug_u = (const float*)d_in[2];   // [8,128,7]
    const float* noise = (const float*)d_in[3];   // [8,128,224,224]
    float*       out   = (float*)d_out;           // [8,256,224,224]

    char* ws = (char*)d_ws;
    float*    prm    = (float*)ws;                                    // 64 KB
    _Float16* wbh    = (_Float16*)(ws + (1 << 16));                   // 1 MB
    _Float16* wav    = wbh + (size_t)NIMG * 512;                      // 1 MB
    _Float16* warped = wav + (size_t)NIMG * 512;                      // 100.4 MB
    _Float16* hbl    = warped + (size_t)NIMG * HWSZ;                  // 100.4 MB

    // 1) per-image params + prebaked WMMA band matrices
    params_kernel<<<NIMG, 32, 0, stream>>>(aug_u, prm, wbh, wav);

    // 2) copy all channels through (augmented ones overwritten later)
    int total4 = (8 * CCH * HWSZ) / 4;   // 25,690,112 float4
    copy_kernel<<<8192, 256, 0, stream>>>((const float4*)M, (float4*)out, total4);

    // 3) warp: 196 blocks/image
    warp_kernel<<<NIMG * 196, 256, 0, stream>>>(M, cidx, prm, warped);

    // 4/5) blur: 196 tiles/image = 28 blocks x 7 waves per image
    hblur_kernel<<<NIMG * 28, 224, 0, stream>>>(warped, wbh, hbl);
    vblur_kernel<<<NIMG * 28, 224, 0, stream>>>(hbl, wav, cidx, noise, out);
}
